// RTNAutoGPTQLinear_82540681494871
// MI455X (gfx1250) — matrix operand order repaired and checked by reading, not compiled, not run on hardware
//
#include <hip/hip_runtime.h>

typedef __attribute__((ext_vector_type(16))) _Float16 v16h;
typedef __attribute__((ext_vector_type(8)))  _Float16 v8h;
typedef __attribute__((ext_vector_type(8)))  float    v8f;
typedef __attribute__((ext_vector_type(4)))  int      v4i;

#define IN_F    4096
#define OUT_F   11008
#define TOKENS  8
#define GROUPS  32              // groups per output row (4096/128)
#define GSIZE   128
#define KSEG    4               // K split across waves
#define SEG_LEN (IN_F / KSEG)   // 1024
#define GPS     (SEG_LEN / GSIZE) // 8 groups per segment

// Stage 0: x [8,4096] fp32 -> zero-padded [16,4096] f16 (A operand staging).
__global__ __launch_bounds__(256) void xpad_f16_kernel(const float* __restrict__ x,
                                                       _Float16* __restrict__ xp) {
    int t = blockIdx.x * 256 + threadIdx.x;     // 0 .. 16*4096-1
    int row = t >> 12;
    int col = t & (IN_F - 1);
    float v = (row < TOKENS) ? x[row * IN_F + col] : 0.0f;
    xp[t] = (_Float16)v;
}

// Main: streaming dequant-GEMM via v_wmma_f32_16x16x32_f16.
// Block = 8 waves: wave w -> (out-tile w&1 of 16 features, K-segment w>>1 of 1024).
__global__ __launch_bounds__(256) void rtn_gemm_wmma(const _Float16* __restrict__ xp,
                                                     const int*      __restrict__ qw,
                                                     const float*    __restrict__ sc,
                                                     float*          __restrict__ out) {
    __shared__ float red[8][8][32];   // [wave][C-vgpr r][lane], conflict-free

    const int wave = threadIdx.x >> 5;
    const int lane = threadIdx.x & 31;
    const int tile = wave & 1;
    const int kseg = wave >> 1;
    const int obase = blockIdx.x * 32 + tile * 16;

    const int n  = lane & 15;        // A row M / B column N / C column N
    const int hi = lane >> 4;
    const int o  = obase + n;        // this lane's output feature

    // A: lane L holds M=L&15; halves 0-15 -> 16 consecutive K starting at hi*16
    const _Float16* arow = xp + (size_t)n * IN_F;
    const int akb = hi * 16;
    // B: lane L holds N=L&15; halves 0-15 -> 16 consecutive K starting at hi*16
    const int* brow = qw + (size_t)o * IN_F;
    const int bkb = hi * 16;

    float acc[8];
#pragma unroll
    for (int r = 0; r < 8; ++r) acc[r] = 0.0f;

    const int segbase = kseg * SEG_LEN;
    for (int g = 0; g < GPS; ++g) {
        const float s = sc[(size_t)o * GROUPS + kseg * GPS + g];
        v8f c = {};
#pragma unroll
        for (int kk = 0; kk < 4; ++kk) {            // 4 x K=32 per 128-group
            const int k0 = segbase + g * GSIZE + kk * 32;

            // ---- A fragment: two 16B loads of staged f16 x (L2-resident, tiny)
            v8h a_lo = *(const v8h*)(arow + k0 + akb);
            v8h a_hi = *(const v8h*)(arow + k0 + akb + 8);
            v16h a;
#pragma unroll
            for (int h = 0; h < 8; ++h) { a[h] = a_lo[h]; a[h + 8] = a_hi[h]; }

            // ---- B fragment: 64 contiguous bytes of qweight, non-temporal b128
            const int* bp = brow + k0 + bkb;
            v4i q0 = __builtin_nontemporal_load((const v4i*)(bp + 0));
            v4i q1 = __builtin_nontemporal_load((const v4i*)(bp + 4));
            v4i q2 = __builtin_nontemporal_load((const v4i*)(bp + 8));
            v4i q3 = __builtin_nontemporal_load((const v4i*)(bp + 12));
            v16h b;
#pragma unroll
            for (int j = 0; j < 4; ++j) {
                b[j]      = (_Float16)(q0[j] - 8);   // exact in f16
                b[4 + j]  = (_Float16)(q1[j] - 8);
                b[8 + j]  = (_Float16)(q2[j] - 8);
                b[12 + j] = (_Float16)(q3[j] - 8);
            }

            c = __builtin_amdgcn_wmma_f32_16x16x32_f16(
                    false, a, false, b, (short)0, c, false, false);
        }
        // apply per-group scale in fp32
#pragma unroll
        for (int r = 0; r < 8; ++r) acc[r] += s * c[r];
    }

    // ---- cross-wave K-segment reduction through LDS
#pragma unroll
    for (int r = 0; r < 8; ++r) red[wave][r][lane] = acc[r];
    __syncthreads();

    if (wave < 2) {  // wave 0 reduces tile 0, wave 1 reduces tile 1
#pragma unroll
        for (int r = 0; r < 8; ++r) {
            float sum = red[wave][r][lane] + red[wave + 2][r][lane]
                      + red[wave + 4][r][lane] + red[wave + 6][r][lane];
            // C layout: lanes 0-15 hold M=r (valid tokens 0..7); lanes 16-31 hold M=r+8 (padding)
            if (lane < 16) out[(size_t)r * OUT_F + obase + lane] = sum;
        }
    }
}

extern "C" void kernel_launch(void* const* d_in, const int* in_sizes, int n_in,
                              void* d_out, int out_size, void* d_ws, size_t ws_size,
                              hipStream_t stream) {
    const float* x  = (const float*)d_in[0];   // [8, 4096] fp32
    const int*   qw = (const int*)d_in[1];     // [11008, 4096] int32 (4-bit values)
    const float* sc = (const float*)d_in[2];   // [11008, 32] fp32
    float* out = (float*)d_out;                // [8, 11008] fp32
    _Float16* xp = (_Float16*)d_ws;            // 16*4096 f16 = 128 KB staging

    xpad_f16_kernel<<<(16 * IN_F) / 256, 256, 0, stream>>>(x, xp);
    rtn_gemm_wmma<<<OUT_F / 32, 256, 0, stream>>>(xp, qw, sc, out);
}
